// MultiExpertLoRALinear_3994319585535
// MI455X (gfx1250) — matrix operand order repaired and checked by reading, not compiled, not run on hardware
//
#include <hip/hip_runtime.h>
#include <math.h>

// Problem dims (from reference setup_inputs)
#define BT   8192     // B*T tokens (M)
#define HH   4096     // hidden (K)
#define OO   4096     // out (N)
#define EACT 4
#define RNK  8
#define KTAIL 32      // EACT*RNK
#define SCALING 2.0f  // 16/8
#define EPSR 1e-12f

typedef __attribute__((ext_vector_type(16))) __bf16 v16bf;
typedef __attribute__((ext_vector_type(8)))  __bf16 v8bf;
typedef __attribute__((ext_vector_type(8)))  float  v8f;
typedef __attribute__((ext_vector_type(4)))  float  v4f;
typedef __attribute__((ext_vector_type(4)))  unsigned int u32x4;
typedef __attribute__((ext_vector_type(8)))  int  i32x8;
typedef __attribute__((ext_vector_type(4)))  int  i32x4;

__device__ __forceinline__ v16bf cat8(v8bf lo, v8bf hi) {
    return __builtin_shufflevector(lo, hi, 0,1,2,3,4,5,6,7,8,9,10,11,12,13,14,15);
}

// ---------------------------------------------------------------------------
// Kernel 1: per-token router logits + z = x@A^T, softmax with prior,
// emit zw[t][e*8+r] = SCALING * alpha[t][e] * z[t][e][r]  (bf16, [BT,32])
// ---------------------------------------------------------------------------
__global__ __launch_bounds__(256) void k_router_zw(
    const float* __restrict__ x,
    const float* __restrict__ lora_A,    // [16,8,H]
    const float* __restrict__ router_W,  // [16,H]
    const float* __restrict__ router_b,  // [16]
    const float* __restrict__ prior,     // [16]
    const int*   __restrict__ active,    // [4]
    __bf16*      __restrict__ zw)        // [BT,32]
{
    const int t    = blockIdx.x;
    const int tid  = threadIdx.x;
    const int lane = tid & 31;
    const int wid  = tid >> 5;

    __shared__ const float* wtab[36];
    __shared__ float part[36][8];
    __shared__ float sres[36];
    __shared__ float salpha[EACT];

    if (tid < 36) {
        const float* p;
        if (tid < EACT) {
            p = router_W + (size_t)active[tid] * HH;
        } else {
            int j = tid - EACT;
            p = lora_A + ((size_t)active[j >> 3] * RNK + (j & 7)) * HH;
        }
        wtab[tid] = p;
    }
    __syncthreads();

    float xi[16];
    const float* xrow = x + (size_t)t * HH;
    #pragma unroll
    for (int j = 0; j < 16; ++j) xi[j] = xrow[j * 256 + tid];

    for (int c = 0; c < 36; ++c) {
        const float* w = wtab[c];
        float p = 0.f;
        #pragma unroll
        for (int j = 0; j < 16; ++j) p += xi[j] * w[j * 256 + tid];
        #pragma unroll
        for (int off = 16; off > 0; off >>= 1) p += __shfl_xor(p, off, 32);
        if (lane == 0) part[c][wid] = p;
    }
    __syncthreads();

    if (tid < 36) {
        float s = 0.f;
        #pragma unroll
        for (int k = 0; k < 8; ++k) s += part[tid][k];
        sres[tid] = s;
    }
    __syncthreads();

    if (tid == 0) {
        float sc[EACT], mx = -1e30f;
        #pragma unroll
        for (int e = 0; e < EACT; ++e) {
            int a = active[e];
            sc[e] = logf(prior[a] + EPSR) + (sres[e] + router_b[a]); // GAMMA=1
            mx = fmaxf(mx, sc[e]);
        }
        float den = 0.f;
        #pragma unroll
        for (int e = 0; e < EACT; ++e) { sc[e] = expf(sc[e] - mx); den += sc[e]; }
        float inv = SCALING / den;
        #pragma unroll
        for (int e = 0; e < EACT; ++e) salpha[e] = sc[e] * inv;
    }
    __syncthreads();

    if (tid < KTAIL) {
        int e = tid >> 3;
        zw[(size_t)t * KTAIL + tid] = (__bf16)(salpha[e] * sres[EACT + tid]);
    }
}

// ---------------------------------------------------------------------------
// Kernel 2: pack lora_B of the active experts as Bcat[o][j] bf16, j = e*8+r
// ---------------------------------------------------------------------------
__global__ __launch_bounds__(256) void k_pack_B(
    const float* __restrict__ lora_B,   // [16,O,8]
    const int*   __restrict__ active,
    __bf16*      __restrict__ Bcat)     // [O,32]
{
    int idx = blockIdx.x * 256 + threadIdx.x;
    int o = idx >> 5, j = idx & 31;
    int a = active[j >> 3], rr = j & 7;
    Bcat[idx] = (__bf16)(lora_B[((size_t)a * OO + o) * RNK + rr]);
}

// ---------------------------------------------------------------------------
// Kernel 2b: bulk fp32 -> bf16 conversion (pre-pass for the TDM GEMM path)
// ---------------------------------------------------------------------------
__global__ __launch_bounds__(256) void k_cvt_bf16(
    const float* __restrict__ src, __bf16* __restrict__ dst, int n8)
{
    int i = blockIdx.x * 256 + threadIdx.x;
    if (i >= n8) return;
    v4f a = *(const v4f*)(src + (size_t)i * 8);
    v4f b = *(const v4f*)(src + (size_t)i * 8 + 4);
    v8bf o;
    #pragma unroll
    for (int j = 0; j < 4; ++j) { o[j] = (__bf16)a[j]; o[j + 4] = (__bf16)b[j]; }
    *(v8bf*)(dst + (size_t)i * 8) = o;
}

// ---------------------------------------------------------------------------
// Shared GEMM pieces
// ---------------------------------------------------------------------------
#define KSTEP 64
#define LSTR  72   // bf16 elems per LDS row: 144B, 16B-aligned, conflict-free
#define NKT   (HH / KSTEP)

__device__ __forceinline__ void wmma_tail_epilogue(
    v8f acc[2][4], const __bf16* __restrict__ zw, const __bf16* __restrict__ Bcat,
    const float* __restrict__ bias, float* __restrict__ out,
    int bm, int bn, int wm, int wn, int ln, int hiHalf, int c0, int kb)
{
    // K-tail (32): LoRA/router contribution, already bf16 in ws
    v16bf afr[2], bfr[4];
    #pragma unroll
    for (int i = 0; i < 2; ++i) {
        const __bf16* ap = zw + (size_t)(bm + wm + i * 16 + ln) * KTAIL + c0;
        afr[i] = cat8(*(const v8bf*)ap, *(const v8bf*)(ap + 16));
    }
    #pragma unroll
    for (int i = 0; i < 4; ++i) {
        const __bf16* bp = Bcat + (size_t)(bn + wn + i * 16 + ln) * KTAIL + c0;
        bfr[i] = cat8(*(const v8bf*)bp, *(const v8bf*)(bp + 16));
    }
    #pragma unroll
    for (int i = 0; i < 2; ++i)
        #pragma unroll
        for (int j = 0; j < 4; ++j)
            acc[i][j] = __builtin_amdgcn_wmma_f32_16x16x32_bf16(
                false, afr[i], false, bfr[j], (short)0, acc[i][j], false, false);

    // epilogue: + bias, store fp32
    #pragma unroll
    for (int i = 0; i < 2; ++i) {
        #pragma unroll
        for (int j = 0; j < 4; ++j) {
            int n = bn + wn + j * 16 + ln;
            float bb = bias[n];
            int mbase = bm + wm + i * 16 + (hiHalf ? 8 : 0);
            #pragma unroll
            for (int v = 0; v < 8; ++v)
                out[(size_t)(mbase + v) * OO + n] = acc[i][j][v] + bb;
        }
    }
}

// ---------------------------------------------------------------------------
// Kernel 3a: TDM GEMM. bf16 x/W already in workspace; tiles move to LDS via
// the Tensor Data Mover (double-buffered, TENSORcnt-tracked), WMMA consumes.
// ---------------------------------------------------------------------------
__device__ __forceinline__ void tdm_load_tile_128x64(
    const __bf16* gsrc, unsigned int lds_off)
{
    // 2-D tile: 128 rows x 64 bf16, row stride HH elems, LDS pad 4 DW / 32 DW
    unsigned long long ga = (unsigned long long)(uintptr_t)gsrc;
    u32x4 g0;
    g0[0] = 1u;                               // count=1, user descriptor
    g0[1] = lds_off;                          // lds_addr (bytes)
    g0[2] = (unsigned int)ga;                 // global_addr[31:0]
    g0[3] = (unsigned int)((ga >> 32) & 0x1FFFFFFull) | (2u << 30); // [56:32], type=2
    const unsigned int TD1 = 1u << 20;        // tensor_dim1 (large, no OOB)
    i32x8 g1;
    g1[0] = (int)((1u << 16) | (1u << 20) | (4u << 22) | (3u << 25));
            // data_size=1(2B), pad_enable, pad_interval=32DW, pad_amount=4DW
    g1[1] = (int)((HH & 0xFFFFu) << 16);                   // tensor_dim0 lo16
    g1[2] = (int)(((HH >> 16) & 0xFFFFu) | ((TD1 & 0xFFFFu) << 16));
    g1[3] = (int)(((TD1 >> 16) & 0xFFFFu) | ((unsigned)KSTEP << 16)); // tile_dim0=64
    g1[4] = 128;                              // tile_dim1=128 rows (tile_dim2=0)
    g1[5] = HH;                               // tensor_dim0_stride lo32
    g1[6] = 0;                                // stride hi16 | dim1_stride lo16
    g1[7] = 0;
    i32x4 z4; z4[0] = 0; z4[1] = 0; z4[2] = 0; z4[3] = 0;
    i32x8 z8; z8[0] = 0; z8[1] = 0; z8[2] = 0; z8[3] = 0;
                z8[4] = 0; z8[5] = 0; z8[6] = 0; z8[7] = 0;
    __builtin_amdgcn_tensor_load_to_lds(g0, g1, z4, z4, z8, 0);
}

__global__ __launch_bounds__(256) void k_gemm_tdm(
    const __bf16* __restrict__ xbf,    // [BT,H] bf16
    const __bf16* __restrict__ wbf,    // [O,H]  bf16
    const float*  __restrict__ bias,
    const __bf16* __restrict__ zw,     // [BT,32]
    const __bf16* __restrict__ Bcat,   // [O,32]
    float*        __restrict__ out)
{
    __shared__ __bf16 Abuf[2][128][LSTR];
    __shared__ __bf16 Bbuf[2][128][LSTR];

    const int tid  = threadIdx.x;
    const int lane = tid & 31;
    const int wid  = tid >> 5;
    const int bm   = blockIdx.y * 128;
    const int bn   = blockIdx.x * 128;
    const int wm   = (wid & 3) * 32;
    const int wn   = (wid >> 2) * 64;

    const int ln     = lane & 15;
    const int hiHalf = lane >> 4;
    const int c0     = hiHalf * 8;
    const int kb     = hiHalf * 16;

    v8f acc[2][4] = {};

    const __bf16* asrc = xbf + (size_t)bm * HH;
    const __bf16* bsrc = wbf + (size_t)bn * HH;

    if (wid == 0) {   // wave 0 drives the TDM (EXEC-independent, wave-uniform)
        tdm_load_tile_128x64(asrc, (unsigned int)(uintptr_t)&Abuf[0][0][0]);
        tdm_load_tile_128x64(bsrc, (unsigned int)(uintptr_t)&Bbuf[0][0][0]);
    }

    for (int it = 0; it < NKT; ++it) {
        __syncthreads();   // buf[(it+1)&1] no longer being read (iter it-1 done)
        if (wid == 0) {
            if (it + 1 < NKT) {
                int nb = (it + 1) & 1;
                tdm_load_tile_128x64(asrc + (size_t)(it + 1) * KSTEP,
                                     (unsigned int)(uintptr_t)&Abuf[nb][0][0]);
                tdm_load_tile_128x64(bsrc + (size_t)(it + 1) * KSTEP,
                                     (unsigned int)(uintptr_t)&Bbuf[nb][0][0]);
                __builtin_amdgcn_s_wait_tensorcnt(2);  // pair "it" complete
            } else {
                __builtin_amdgcn_s_wait_tensorcnt(0);
            }
        }
        __syncthreads();   // tile "it" visible to all waves

        const __bf16 (*As)[LSTR] = Abuf[it & 1];
        const __bf16 (*Bs)[LSTR] = Bbuf[it & 1];
        #pragma unroll
        for (int kc = 0; kc < KSTEP; kc += 32) {
            v16bf afr[2], bfr[4];
            #pragma unroll
            for (int i = 0; i < 2; ++i) {
                const __bf16* ap = &As[wm + i * 16 + ln][kc + c0];
                afr[i] = cat8(*(const v8bf*)ap, *(const v8bf*)(ap + 16));
            }
            #pragma unroll
            for (int i = 0; i < 4; ++i) {
                const __bf16* bp = &Bs[wn + i * 16 + ln][kc + c0];
                bfr[i] = cat8(*(const v8bf*)bp, *(const v8bf*)(bp + 16));
            }
            #pragma unroll
            for (int i = 0; i < 2; ++i)
                #pragma unroll
                for (int j = 0; j < 4; ++j)
                    acc[i][j] = __builtin_amdgcn_wmma_f32_16x16x32_bf16(
                        false, afr[i], false, bfr[j], (short)0, acc[i][j], false, false);
        }
    }

    wmma_tail_epilogue(acc, zw, Bcat, bias, out, bm, bn, wm, wn, ln, hiHalf, c0, kb);
}

// ---------------------------------------------------------------------------
// Kernel 3b: fallback GEMM (fp32 in, register-staged cvt, software pipeline)
// used when the workspace cannot hold the bf16 copies.
// ---------------------------------------------------------------------------
__global__ __launch_bounds__(256) void k_gemm(
    const float*  __restrict__ x,
    const float*  __restrict__ W,
    const float*  __restrict__ bias,
    const __bf16* __restrict__ zw,
    const __bf16* __restrict__ Bcat,
    float*        __restrict__ out)
{
    __shared__ __bf16 As[128][LSTR];
    __shared__ __bf16 Bs[128][LSTR];

    const int tid  = threadIdx.x;
    const int lane = tid & 31;
    const int wid  = tid >> 5;
    const int bm   = blockIdx.y * 128;
    const int bn   = blockIdx.x * 128;
    const int wm   = (wid & 3) * 32;
    const int wn   = (wid >> 2) * 64;

    const int row  = tid >> 1;
    const int half = (tid & 1) * 32;

    const int ln     = lane & 15;
    const int hiHalf = lane >> 4;
    const int c0     = hiHalf * 8;
    const int kb     = hiHalf * 16;

    v8f acc[2][4] = {};

    const float* xsrc = x + (size_t)(bm + row) * HH + half;
    const float* wsrc = W + (size_t)(bn + row) * HH + half;

    v4f xr[8], wr[8];
    #pragma unroll
    for (int i = 0; i < 8; ++i) xr[i] = *(const v4f*)(xsrc + 4 * i);
    #pragma unroll
    for (int i = 0; i < 8; ++i) wr[i] = *(const v4f*)(wsrc + 4 * i);

    for (int k0 = 0; k0 < HH; k0 += KSTEP) {
        #pragma unroll
        for (int s = 0; s < 4; ++s) {
            v8bf xb, wb;
            #pragma unroll
            for (int i = 0; i < 4; ++i) {
                xb[i]     = (__bf16)xr[2 * s][i];
                xb[i + 4] = (__bf16)xr[2 * s + 1][i];
                wb[i]     = (__bf16)wr[2 * s][i];
                wb[i + 4] = (__bf16)wr[2 * s + 1][i];
            }
            *(v8bf*)&As[row][half + 8 * s] = xb;
            *(v8bf*)&Bs[row][half + 8 * s] = wb;
        }
        __syncthreads();

        if (k0 + KSTEP < HH) {
            const float* xp = xsrc + k0 + KSTEP;
            const float* wp = wsrc + k0 + KSTEP;
            #pragma unroll
            for (int i = 0; i < 8; ++i) xr[i] = *(const v4f*)(xp + 4 * i);
            #pragma unroll
            for (int i = 0; i < 8; ++i) wr[i] = *(const v4f*)(wp + 4 * i);
            if (k0 + 2 * KSTEP < HH) {
                __builtin_prefetch(xp + KSTEP, 0, 1);
                __builtin_prefetch(wp + KSTEP, 0, 1);
            }
        }

        #pragma unroll
        for (int kc = 0; kc < KSTEP; kc += 32) {
            v16bf afr[2], bfr[4];
            #pragma unroll
            for (int i = 0; i < 2; ++i) {
                const __bf16* ap = &As[wm + i * 16 + ln][kc + c0];
                afr[i] = cat8(*(const v8bf*)ap, *(const v8bf*)(ap + 16));
            }
            #pragma unroll
            for (int i = 0; i < 4; ++i) {
                const __bf16* bp = &Bs[wn + i * 16 + ln][kc + c0];
                bfr[i] = cat8(*(const v8bf*)bp, *(const v8bf*)(bp + 16));
            }
            #pragma unroll
            for (int i = 0; i < 2; ++i)
                #pragma unroll
                for (int j = 0; j < 4; ++j)
                    acc[i][j] = __builtin_amdgcn_wmma_f32_16x16x32_bf16(
                        false, afr[i], false, bfr[j], (short)0, acc[i][j], false, false);
        }
        __syncthreads();
    }

    wmma_tail_epilogue(acc, zw, Bcat, bias, out, bm, bn, wm, wn, ln, hiHalf, c0, kb);
}

// ---------------------------------------------------------------------------
extern "C" void kernel_launch(void* const* d_in, const int* in_sizes, int n_in,
                              void* d_out, int out_size, void* d_ws, size_t ws_size,
                              hipStream_t stream) {
    const float* x      = (const float*)d_in[0];
    const float* W      = (const float*)d_in[1];
    const float* b      = (const float*)d_in[2];
    const float* lora_A = (const float*)d_in[3];
    const float* lora_B = (const float*)d_in[4];
    const float* rW     = (const float*)d_in[5];
    const float* rb     = (const float*)d_in[6];
    const float* prior  = (const float*)d_in[7];
    const int*   active = (const int*)d_in[8];
    float* out = (float*)d_out;

    char* wsb = (char*)d_ws;
    __bf16* zw   = (__bf16*)wsb;                                  // 512 KB
    __bf16* Bcat = (__bf16*)(wsb + (size_t)BT * KTAIL * 2);       // 256 KB
    __bf16* xbf  = (__bf16*)(wsb + (size_t)(BT + OO) * KTAIL * 2);
    __bf16* wbf  = xbf + (size_t)BT * HH;
    const size_t need = (size_t)(BT + OO) * KTAIL * 2 + (size_t)(BT + OO) * HH * 2;

    k_router_zw<<<BT, 256, 0, stream>>>(x, lora_A, rW, rb, prior, active, zw);
    k_pack_B<<<(OO * KTAIL) / 256, 256, 0, stream>>>(lora_B, active, Bcat);

    dim3 grid(OO / 128, BT / 128);
    if (ws_size >= need) {
        k_cvt_bf16<<<(BT * HH / 8 + 255) / 256, 256, 0, stream>>>(x, xbf, BT * HH / 8);
        k_cvt_bf16<<<(OO * HH / 8 + 255) / 256, 256, 0, stream>>>(W, wbf, OO * HH / 8);
        k_gemm_tdm<<<grid, 256, 0, stream>>>(xbf, wbf, b, zw, Bcat, out);
    } else {
        k_gemm<<<grid, 256, 0, stream>>>(x, W, b, zw, Bcat, out);
    }
}
